// TinyViTBlock_62672162783291
// MI455X (gfx1250) — compile-verified, matrix operand order checked by reading, not run on hardware
//
#include <hip/hip_runtime.h>
#include <math.h>

typedef __bf16 bf16;
typedef unsigned int u32;
typedef __attribute__((ext_vector_type(16))) __bf16 v16bf;
typedef __attribute__((ext_vector_type(8)))  float  v8f;
typedef __attribute__((ext_vector_type(4)))  unsigned int v4u;
typedef __attribute__((ext_vector_type(8)))  unsigned int v8u;

#define WS_WIN 14
#define NTOK   196
#define MPAD   208   // 13 tiles of 16
#define KPAD   224   // 7 tiles of 32 for the PV K dimension
#define CCH    384
#define NHEADS 12
#define KDIM   32
#define NWIN   200
#define HIDF   1536
#define NBATCH 8
#define IMGHW  64
#define SCALE_ATTN 0.17677669529663689f

// ---- workspace offsets (bytes) ----
#define OFF_XN    ((size_t)0)            // bf16 [200][208][384]   31,948,800 B
#define OFF_Q     ((size_t)31948800)     // bf16 [200*12][208][32]
#define OFF_K     ((size_t)63897600)
#define OFF_V     ((size_t)95846400)
#define OFF_AO    ((size_t)0)            // alias XN (dead after QKV gemm)
#define OFF_H1    ((size_t)0)            // alias XN+Q+K+part of V (dead after attention)
#define OFF_XATTN ((size_t)127795200)    // f32 [8][4096][384]
#define OFF_X2    ((size_t)178126848)    // f32 [8][4096][384]
#define OFF_XNMLP ((size_t)228458496)    // bf16 [32768][384]
#define OFF_WQKV  ((size_t)253624320)    // bf16 [1152][384]
#define OFF_WPROJ ((size_t)254509056)    // bf16 [384][384]
#define OFF_WFC1  ((size_t)254803968)    // bf16 [1536][384]
#define OFF_WFC2  ((size_t)255983616)    // bf16 [384][1536]

static __device__ inline v8f vzero8() {
  v8f z = {0.f,0.f,0.f,0.f,0.f,0.f,0.f,0.f};
  return z;
}

static __device__ inline v8f wmma_bf16(v16bf a, v16bf b, v8f c) {
  // (neg_a, A, neg_b, B, c_mod, C, reuse_a, reuse_b)
  return __builtin_amdgcn_wmma_f32_16x16x32_bf16(false, a, false, b, (short)0, c,
                                                 false, false);
}

// ---- Tensor Data Mover: contiguous global -> LDS copy as a 1-D tile ----
// nbytes must be a multiple of 8 and nbytes/8 <= 65535.
// Call from exactly ONE wave (TDM ignores EXEC; one issue per tile), then
// s_wait_tensorcnt(0) in that wave before the block barrier.
static __device__ inline void tdm_load_to_lds(void* lds_dst, const void* gsrc,
                                              u32 nbytes) {
  u32 lds_off = (u32)(size_t)lds_dst;   // generic LDS addr: low 32 bits = offset
  unsigned long long ga = (unsigned long long)(size_t)gsrc;
  u32 n8 = nbytes >> 3;                 // elements of data_size = 8 bytes
  v4u g0;
  g0[0] = 1u;                            // count=1, user descriptor
  g0[1] = lds_off;                       // lds_addr
  g0[2] = (u32)ga;                       // global_addr[31:0]
  g0[3] = ((u32)(ga >> 32) & 0x01FFFFFFu) | (2u << 30);  // addr[56:32], type=2
  v8u g1;
  g1[0] = (3u << 16);                    // workgroup_mask=0, data_size=3 (8B)
  g1[1] = (n8 & 0xFFFFu) << 16;          // tensor_dim0[15:0]
  g1[2] = (n8 >> 16) | (1u << 16);       // tensor_dim0[31:16], tensor_dim1=1
  g1[3] = (n8 & 0xFFFFu) << 16;          // tensor_dim1 hi=0, tile_dim0=n8
  g1[4] = 0u;                            // tile_dim1=0, tile_dim2=0
  g1[5] = n8;                            // tensor_dim0_stride[31:0]
  g1[6] = 0u;                            // stride0 hi, stride1 lo
  g1[7] = 0u;                            // stride1 hi
  v4u g2;
  g2[0] = 1u;                            // tensor_dim2 = 1
  g2[1] = 1u;                            // tensor_dim3 = 1 (iterate disabled)
  g2[2] = n8;                            // tensor_dim2_stride[31:0]
  g2[3] = 0u;                            // stride hi, tile_dim3 = 0
  v4u g3;
  g3[0] = 0u; g3[1] = 0u; g3[2] = 0u; g3[3] = 0u;
  asm volatile("tensor_load_to_lds %0, %1, %2, %3"
               :: "s"(g0), "s"(g1), "s"(g2), "s"(g3)
               : "memory");
}

// A fragment 16x32 bf16, row-major, elem (m,k) at p[m*ld + k].
// ISA layout: lanes 0-15 -> M=lane, K groups {0..7,16..23}; lanes 16-31 -> K+8.
// Also serves as B^T loader (weights stored [N][K] row-major): row index = N.
static __device__ inline v16bf frag_a(const bf16* __restrict__ p, int ld,
                                      int kbase, int lane) {
  int m  = lane & 15;
  int kh = (lane >> 4) << 3;
  const bf16* row = p + (size_t)m * ld + kbase;
  v16bf f;
#pragma unroll
  for (int v = 0; v < 8; ++v) {
    int k = ((v >> 2) << 4) + kh + ((v & 3) << 1);
    f[2*v]     = row[k];
    f[2*v + 1] = row[k + 1];
  }
  return f;
}

// B fragment 32x16 bf16, row-major KxN, elem (k,n) at p[(kbase+k)*ld + noff+n].
static __device__ inline v16bf frag_b(const bf16* __restrict__ p, int ld,
                                      int kbase, int noff, int lane) {
  int n  = (lane & 15) + noff;
  int kh = (lane >> 4) << 3;
  v16bf f;
#pragma unroll
  for (int v = 0; v < 8; ++v) {
    int k = kbase + ((v >> 2) << 4) + kh + ((v & 3) << 1);
    f[2*v]     = p[(size_t)k * ld + n];
    f[2*v + 1] = p[(size_t)(k + 1) * ld + n];
  }
  return f;
}

// ---------------- weight f32 -> bf16 ----------------
__global__ void cvt_bf16_kernel(const float* __restrict__ a, bf16* __restrict__ o,
                                int n) {
  int i = blockIdx.x * 256 + threadIdx.x;
  if (i < n) o[i] = (bf16)a[i];
}

// ---------------- stage 1: window partition + LN -> bf16 ----------------
__global__ void ln_window_kernel(const float* __restrict__ x,
                                 const float* __restrict__ w,
                                 const float* __restrict__ b,
                                 bf16* __restrict__ xn) {
  int win = blockIdx.x;                       // 0..199
  int bimg = win / 25, wy = (win % 25) / 5, wx = win % 5;
  int lane = threadIdx.x & 31, wv = threadIdx.x >> 5;
  for (int t = wv; t < MPAD; t += 8) {
    bf16* dst = xn + ((size_t)win * MPAD + t) * CCH;
    if (t >= NTOK) {                          // pure GEMM padding rows
      for (int c = lane; c < CCH; c += 32) dst[c] = (bf16)0.f;
      continue;
    }
    int ty = t / WS_WIN, tx = t % WS_WIN;
    int y = wy * WS_WIN + ty, xc = wx * WS_WIN + tx;
    bool valid = (y < IMGHW) && (xc < IMGHW);
    const float* src = x + (((size_t)bimg * 4096) + (size_t)y * IMGHW + xc) * CCH;
    float vals[12];
    float s = 0.f;
#pragma unroll
    for (int i = 0; i < 12; ++i) {
      float v2 = valid ? src[lane + 32 * i] : 0.f;
      vals[i] = v2; s += v2;
    }
    for (int off = 16; off > 0; off >>= 1) s += __shfl_xor(s, off, 32);
    float mean = s * (1.f / CCH);
    float vs = 0.f;
#pragma unroll
    for (int i = 0; i < 12; ++i) { float d = vals[i] - mean; vs += d * d; }
    for (int off = 16; off > 0; off >>= 1) vs += __shfl_xor(vs, off, 32);
    float inv = rsqrtf(vs * (1.f / CCH) + 1e-5f);
#pragma unroll
    for (int i = 0; i < 12; ++i) {
      int c = lane + 32 * i;
      dst[c] = (bf16)((vals[i] - mean) * inv * w[c] + b[c]);
    }
  }
}

// ---------------- stage 2: QKV GEMM (per window 208 x 1152 x 384) ----------------
__global__ void qkv_gemm_kernel(const bf16* __restrict__ xn,
                                const bf16* __restrict__ wq,
                                const float* __restrict__ qkv_b,
                                bf16* __restrict__ q, bf16* __restrict__ k,
                                bf16* __restrict__ v) {
  __shared__ bf16 As[16 * CCH];
  int wm = blockIdx.x, win = wm / 13, mt = wm % 13;
  int ng = blockIdx.y;
  int lane = threadIdx.x & 31, wv = threadIdx.x >> 5;
  const bf16* Ag = xn + ((size_t)win * MPAD + mt * 16) * CCH;
  if (threadIdx.x < 32) {
    tdm_load_to_lds(As, Ag, 16 * CCH * 2);
    __builtin_amdgcn_s_wait_tensorcnt(0);
  }
  __syncthreads();

  int nt = ng * 8 + wv;           // 0..71
  int o0 = nt * 16;
  const bf16* wpan = wq + (size_t)o0 * CCH;
  __builtin_prefetch(wpan, 0, 0);
  v8f acc = vzero8();
#pragma unroll
  for (int kk = 0; kk < 12; ++kk) {
    v16bf a  = frag_a(As, CCH, kk * 32, lane);
    v16bf bf = frag_a(wpan, CCH, kk * 32, lane);  // B^T
    acc = wmma_bf16(a, bf, acc);
  }
  int o = o0 + (lane & 15);
  int head = o / 96, part = (o % 96) / 32, d = o % 32;
  bf16* outp = (part == 0) ? q : (part == 1) ? k : v;
  float bi = qkv_b[o];
  size_t basep = (((size_t)win * NHEADS + head) * MPAD + mt * 16) * KDIM + d;
  int rbase = (lane >> 4) * 8;
#pragma unroll
  for (int r = 0; r < 8; ++r) {
    int m = rbase + r;
    outp[basep + (size_t)m * KDIM] = (bf16)(acc[r] + bi);
  }
}

// ---------------- stage 3: attention per (window, head) ----------------
__global__ void attn_kernel(const bf16* __restrict__ q, const bf16* __restrict__ k,
                            const bf16* __restrict__ v,
                            const float* __restrict__ biases,
                            const int* __restrict__ bidx,
                            bf16* __restrict__ attn_out) {
  __shared__ bf16 Ks[KPAD * KDIM];
  __shared__ bf16 Vs[KPAD * KDIM];
  __shared__ bf16 Ps[8 * 16 * KPAD];      // wave-private P strips
  int wh = blockIdx.x;                    // 0..2399
  int win = wh / NHEADS, h = wh % NHEADS;
  size_t base = (size_t)wh * MPAD * KDIM;
  // zero K-dimension padding rows (208..223) of Ks/Vs
  for (int i = threadIdx.x + MPAD * KDIM; i < KPAD * KDIM; i += 256) {
    Ks[i] = (bf16)0.f; Vs[i] = (bf16)0.f;
  }
  if (threadIdx.x < 32) {
    tdm_load_to_lds(Ks, k + base, MPAD * KDIM * 2);
    tdm_load_to_lds(Vs, v + base, MPAD * KDIM * 2);
    __builtin_amdgcn_s_wait_tensorcnt(0);
  }
  __syncthreads();
  int lane = threadIdx.x & 31, wv = threadIdx.x >> 5;
  bf16* Pw = &Ps[wv * 16 * KPAD];
  // zero K-padding columns of this wave's strip once (never rewritten)
  for (int idx = lane; idx < 16 * 16; idx += 32) {
    int r = idx >> 4, c = MPAD + (idx & 15);
    Pw[r * KPAD + c] = (bf16)0.f;
  }
  int col = lane & 15, rbase = (lane >> 4) * 8;

  for (int mt = wv; mt < 13; mt += 8) {
    v16bf qf = frag_a(q + base + (size_t)mt * 16 * KDIM, KDIM, 0, lane);
    v8f S[13];
#pragma unroll
    for (int nt = 0; nt < 13; ++nt) {
      v16bf kf = frag_a(Ks + (size_t)nt * 16 * KDIM, KDIM, 0, lane);  // k^T
      S[nt] = wmma_bf16(qf, kf, vzero8());
    }
    // scale + relative position bias + mask + softmax (rows live in C layout)
#pragma unroll
    for (int r = 0; r < 8; ++r) {
      int mrow = mt * 16 + rbase + r;
      float mx = -1e30f;
#pragma unroll
      for (int nt = 0; nt < 13; ++nt) {
        int j = nt * 16 + col;
        float sc = S[nt][r] * SCALE_ATTN;
        if (j < NTOK && mrow < NTOK) sc += biases[h * NTOK + bidx[mrow * NTOK + j]];
        if (j >= NTOK) sc = -1e30f;
        S[nt][r] = sc;
        mx = fmaxf(mx, sc);
      }
      mx = fmaxf(mx, __shfl_xor(mx, 1, 32));
      mx = fmaxf(mx, __shfl_xor(mx, 2, 32));
      mx = fmaxf(mx, __shfl_xor(mx, 4, 32));
      mx = fmaxf(mx, __shfl_xor(mx, 8, 32));
      float sum = 0.f;
#pragma unroll
      for (int nt = 0; nt < 13; ++nt) {
        float e = __expf(S[nt][r] - mx);
        S[nt][r] = e; sum += e;
      }
      sum += __shfl_xor(sum, 1, 32);
      sum += __shfl_xor(sum, 2, 32);
      sum += __shfl_xor(sum, 4, 32);
      sum += __shfl_xor(sum, 8, 32);
      float inv = 1.f / sum;
#pragma unroll
      for (int nt = 0; nt < 13; ++nt)
        Pw[(rbase + r) * KPAD + nt * 16 + col] = (bf16)(S[nt][r] * inv);
    }
    // PV: out(16x32) = P(16x224) x V(224x32)
    v8f o0 = vzero8(), o1 = vzero8();
#pragma unroll
    for (int kt = 0; kt < 7; ++kt) {
      v16bf pf  = frag_a(Pw, KPAD, kt * 32, lane);
      v16bf vf0 = frag_b(Vs, KDIM, kt * 32, 0, lane);
      v16bf vf1 = frag_b(Vs, KDIM, kt * 32, 16, lane);
      o0 = wmma_bf16(pf, vf0, o0);
      o1 = wmma_bf16(pf, vf1, o1);
    }
    size_t ob = ((size_t)win * MPAD + mt * 16) * CCH + h * KDIM;
#pragma unroll
    for (int r = 0; r < 8; ++r) {
      int m = rbase + r;
      attn_out[ob + (size_t)m * CCH + col]      = (bf16)o0[r];
      attn_out[ob + (size_t)m * CCH + 16 + col] = (bf16)o1[r];
    }
  }
}

// ---------------- stage 4: proj GEMM + residual + un-window ----------------
__global__ void proj_kernel(const bf16* __restrict__ ao, const bf16* __restrict__ wp,
                            const float* __restrict__ pb,
                            const float* __restrict__ x_in,
                            float* __restrict__ x_attn) {
  __shared__ bf16 As[16 * CCH];
  int wm = blockIdx.x, win = wm / 13, mt = wm % 13;
  int ng = blockIdx.y;
  int lane = threadIdx.x & 31, wv = threadIdx.x >> 5;
  const bf16* Ag = ao + ((size_t)win * MPAD + mt * 16) * CCH;
  if (threadIdx.x < 32) {
    tdm_load_to_lds(As, Ag, 16 * CCH * 2);
    __builtin_amdgcn_s_wait_tensorcnt(0);
  }
  __syncthreads();

  int nt = ng * 8 + wv;            // 0..23
  int o0 = nt * 16;
  const bf16* wpan = wp + (size_t)o0 * CCH;
  __builtin_prefetch(wpan, 0, 0);
  v8f acc = vzero8();
#pragma unroll
  for (int kk = 0; kk < 12; ++kk) {
    v16bf a  = frag_a(As, CCH, kk * 32, lane);
    v16bf bf = frag_a(wpan, CCH, kk * 32, lane);
    acc = wmma_bf16(a, bf, acc);
  }
  int o = o0 + (lane & 15);
  int rbase = (lane >> 4) * 8;
  int bimg = win / 25, wy = (win % 25) / 5, wx = win % 5;
  float bi = pb[o];
#pragma unroll
  for (int r = 0; r < 8; ++r) {
    int m = mt * 16 + rbase + r;
    if (m >= NTOK) continue;
    int y = wy * WS_WIN + m / WS_WIN, xc = wx * WS_WIN + m % WS_WIN;
    if (y >= IMGHW || xc >= IMGHW) continue;
    size_t idx = (((size_t)bimg * 4096) + (size_t)y * IMGHW + xc) * CCH + o;
    x_attn[idx] = x_in[idx] + acc[r] + bi;
  }
}

// ---------------- stage 5: depthwise 3x3 conv + BN ----------------
__global__ void conv_bn_kernel(const float* __restrict__ xin,
                               const float* __restrict__ cw,
                               const float* __restrict__ bnw,
                               const float* __restrict__ bnb,
                               const float* __restrict__ bnm,
                               const float* __restrict__ bnv,
                               float* __restrict__ x2) {
  size_t i = (size_t)blockIdx.x * 256 + threadIdx.x;
  const size_t total = (size_t)NBATCH * 4096 * CCH;
  if (i >= total) return;
  int c = (int)(i % CCH);
  size_t t = i / CCH;
  int xp = (int)(t % IMGHW), yp = (int)((t / IMGHW) % IMGHW), bb = (int)(t / 4096);
  float s = 0.f;
#pragma unroll
  for (int dy = 0; dy < 3; ++dy)
#pragma unroll
    for (int dx = 0; dx < 3; ++dx) {
      int yy = yp + dy - 1, xx = xp + dx - 1;
      if (yy >= 0 && yy < IMGHW && xx >= 0 && xx < IMGHW)
        s += cw[c * 9 + dy * 3 + dx] *
             xin[(((size_t)bb * 4096) + (size_t)yy * IMGHW + xx) * CCH + c];
    }
  float inv = rsqrtf(bnv[c] + 1e-5f);
  x2[i] = (s - bnm[c]) * inv * bnw[c] + bnb[c];
}

// ---------------- stage 6: LN (mlp) -> bf16 ----------------
__global__ void ln_mlp_kernel(const float* __restrict__ x,
                              const float* __restrict__ w,
                              const float* __restrict__ b,
                              bf16* __restrict__ xn) {
  int lane = threadIdx.x & 31, wv = threadIdx.x >> 5;
  size_t t = (size_t)blockIdx.x * 8 + wv;    // token id 0..32767
  const float* src = x + t * CCH;
  bf16* dst = xn + t * CCH;
  float vals[12];
  float s = 0.f;
#pragma unroll
  for (int i = 0; i < 12; ++i) { vals[i] = src[lane + 32 * i]; s += vals[i]; }
  for (int off = 16; off > 0; off >>= 1) s += __shfl_xor(s, off, 32);
  float mean = s * (1.f / CCH);
  float vs = 0.f;
#pragma unroll
  for (int i = 0; i < 12; ++i) { float d = vals[i] - mean; vs += d * d; }
  for (int off = 16; off > 0; off >>= 1) vs += __shfl_xor(vs, off, 32);
  float inv = rsqrtf(vs * (1.f / CCH) + 1e-5f);
#pragma unroll
  for (int i = 0; i < 12; ++i) {
    int c = lane + 32 * i;
    dst[c] = (bf16)((vals[i] - mean) * inv * w[c] + b[c]);
  }
}

// ---------------- stage 7: fc1 + GELU ----------------
__global__ void fc1_kernel(const bf16* __restrict__ xn, const bf16* __restrict__ w1,
                           const float* __restrict__ b1, bf16* __restrict__ h1) {
  __shared__ bf16 As[16 * CCH];
  int mt = blockIdx.x;                 // 0..2047
  int ng = blockIdx.y;                 // 0..11
  int lane = threadIdx.x & 31, wv = threadIdx.x >> 5;
  const bf16* Ag = xn + (size_t)mt * 16 * CCH;
  if (threadIdx.x < 32) {
    tdm_load_to_lds(As, Ag, 16 * CCH * 2);
    __builtin_amdgcn_s_wait_tensorcnt(0);
  }
  __syncthreads();

  int o0 = (ng * 8 + wv) * 16;         // 0..1520
  const bf16* wpan = w1 + (size_t)o0 * CCH;
  __builtin_prefetch(wpan, 0, 0);
  v8f acc = vzero8();
#pragma unroll
  for (int kk = 0; kk < 12; ++kk) {
    v16bf a  = frag_a(As, CCH, kk * 32, lane);
    v16bf bf = frag_a(wpan, CCH, kk * 32, lane);
    acc = wmma_bf16(a, bf, acc);
  }
  int o = o0 + (lane & 15);
  float bi = b1[o];
  int rbase = (lane >> 4) * 8;
  size_t ob = ((size_t)mt * 16) * HIDF + o;
#pragma unroll
  for (int r = 0; r < 8; ++r) {
    float xv = acc[r] + bi;
    float g = 0.5f * xv * (1.f + erff(xv * 0.70710678118654752f));
    h1[ob + (size_t)(rbase + r) * HIDF] = (bf16)g;
  }
}

// ---------------- stage 8: fc2 + residual ----------------
__global__ void fc2_kernel(const bf16* __restrict__ h1, const bf16* __restrict__ w2,
                           const float* __restrict__ b2,
                           const float* __restrict__ x2, float* __restrict__ out) {
  __shared__ bf16 As[16 * HIDF];
  int mt = blockIdx.x;                 // 0..2047
  int ng = blockIdx.y;                 // 0..2
  int lane = threadIdx.x & 31, wv = threadIdx.x >> 5;
  const bf16* Ag = h1 + (size_t)mt * 16 * HIDF;
  if (threadIdx.x < 32) {
    tdm_load_to_lds(As, Ag, 16 * HIDF * 2);   // 48 KB in one TDM op
    __builtin_amdgcn_s_wait_tensorcnt(0);
  }
  __syncthreads();

  int o0 = (ng * 8 + wv) * 16;         // 0..368
  const bf16* wpan = w2 + (size_t)o0 * HIDF;
  __builtin_prefetch(wpan, 0, 0);
  v8f acc = vzero8();
#pragma unroll
  for (int kk = 0; kk < 48; ++kk) {
    v16bf a  = frag_a(As, HIDF, kk * 32, lane);
    v16bf bf = frag_a(wpan, HIDF, kk * 32, lane);
    acc = wmma_bf16(a, bf, acc);
  }
  int o = o0 + (lane & 15);
  float bi = b2[o];
  int rbase = (lane >> 4) * 8;
  size_t ob = ((size_t)mt * 16) * CCH + o;
#pragma unroll
  for (int r = 0; r < 8; ++r) {
    size_t idx = ob + (size_t)(rbase + r) * CCH;
    out[idx] = x2[idx] + acc[r] + bi;
  }
}

extern "C" void kernel_launch(void* const* d_in, const int* in_sizes, int n_in,
                              void* d_out, int out_size, void* d_ws, size_t ws_size,
                              hipStream_t stream) {
  const float* x        = (const float*)d_in[0];
  const float* ln_a_w   = (const float*)d_in[1];
  const float* ln_a_b   = (const float*)d_in[2];
  const float* qkv_w    = (const float*)d_in[3];
  const float* qkv_b    = (const float*)d_in[4];
  const float* proj_w   = (const float*)d_in[5];
  const float* proj_b   = (const float*)d_in[6];
  const float* biases   = (const float*)d_in[7];
  const float* conv_w   = (const float*)d_in[8];
  const float* bn_w     = (const float*)d_in[9];
  const float* bn_b     = (const float*)d_in[10];
  const float* bn_mean  = (const float*)d_in[11];
  const float* bn_var   = (const float*)d_in[12];
  const float* ln_m_w   = (const float*)d_in[13];
  const float* ln_m_b   = (const float*)d_in[14];
  const float* fc1_w    = (const float*)d_in[15];
  const float* fc1_b    = (const float*)d_in[16];
  const float* fc2_w    = (const float*)d_in[17];
  const float* fc2_b    = (const float*)d_in[18];
  const int*   bias_idx = (const int*)d_in[19];

  char* ws = (char*)d_ws;
  bf16* xn_win = (bf16*)(ws + OFF_XN);
  bf16* qw     = (bf16*)(ws + OFF_Q);
  bf16* kw     = (bf16*)(ws + OFF_K);
  bf16* vw     = (bf16*)(ws + OFF_V);
  bf16* ao     = (bf16*)(ws + OFF_AO);
  bf16* h1     = (bf16*)(ws + OFF_H1);
  float* xattn = (float*)(ws + OFF_XATTN);
  float* x2    = (float*)(ws + OFF_X2);
  bf16* xnmlp  = (bf16*)(ws + OFF_XNMLP);
  bf16* wqkv   = (bf16*)(ws + OFF_WQKV);
  bf16* wproj  = (bf16*)(ws + OFF_WPROJ);
  bf16* wfc1   = (bf16*)(ws + OFF_WFC1);
  bf16* wfc2   = (bf16*)(ws + OFF_WFC2);

  // weights -> bf16
  cvt_bf16_kernel<<<(442368 + 255) / 256, 256, 0, stream>>>(qkv_w, wqkv, 442368);
  cvt_bf16_kernel<<<(147456 + 255) / 256, 256, 0, stream>>>(proj_w, wproj, 147456);
  cvt_bf16_kernel<<<(589824 + 255) / 256, 256, 0, stream>>>(fc1_w, wfc1, 589824);
  cvt_bf16_kernel<<<(589824 + 255) / 256, 256, 0, stream>>>(fc2_w, wfc2, 589824);

  ln_window_kernel<<<NWIN, 256, 0, stream>>>(x, ln_a_w, ln_a_b, xn_win);
  qkv_gemm_kernel<<<dim3(NWIN * 13, 9), 256, 0, stream>>>(xn_win, wqkv, qkv_b,
                                                          qw, kw, vw);
  attn_kernel<<<NWIN * NHEADS, 256, 0, stream>>>(qw, kw, vw, biases, bias_idx, ao);
  proj_kernel<<<dim3(NWIN * 13, 3), 256, 0, stream>>>(ao, wproj, proj_b, x, xattn);
  conv_bn_kernel<<<49152, 256, 0, stream>>>(xattn, conv_w, bn_w, bn_b, bn_mean,
                                            bn_var, x2);
  ln_mlp_kernel<<<4096, 256, 0, stream>>>(x2, ln_m_w, ln_m_b, xnmlp);
  fc1_kernel<<<dim3(2048, 12), 256, 0, stream>>>(xnmlp, wfc1, fc1_b, h1);
  fc2_kernel<<<dim3(2048, 3), 256, 0, stream>>>(h1, wfc2, fc2_b, x2,
                                                (float*)d_out);
}
